// LieMoE_54503134986835
// MI455X (gfx1250) — matrix operand order repaired and checked by reading, not compiled, not run on hardware
//
#include <hip/hip_runtime.h>
#include <stdint.h>

typedef __bf16 bf16;
typedef __attribute__((ext_vector_type(16))) __bf16 v16bf;
typedef __attribute__((ext_vector_type(8)))  __bf16 v8bf;
typedef __attribute__((ext_vector_type(8)))  float   v8f;
typedef __attribute__((ext_vector_type(4)))  float   v4f;

#define DEVI __device__ __forceinline__

constexpr int N_TOK = 8192;
constexpr int DIM   = 1024;
constexpr int HID   = 4096;
constexpr int OUTD  = 1024;
constexpr int NEXP  = 8;
constexpr int TOPK  = 3;

// ---------------- CDNA5 async global->LDS copy (ASYNCcnt path) ----------------
// Per-lane: LDS[lds + OFF .. +15] = MEM[gaddr + OFF .. +15]  (ISA 15.18.3 op 98)
template <int OFF>
DEVI void async_b128(uint32_t lds, const bf16* g) {
  asm volatile("global_load_async_to_lds_b128 %0, %1, off offset:%2"
               :: "v"(lds), "v"((unsigned long long)(uintptr_t)g), "i"(OFF)
               : "memory");
}
DEVI void wait_async() { asm volatile("s_wait_asynccnt 0x0" ::: "memory"); }
// low 32 bits of a flat LDS pointer == LDS byte offset (aperture rule)
DEVI uint32_t lds_off(const void* p) { return (uint32_t)(uintptr_t)p; }

// stage one thread's share of a K-step tile: two 64B A rows + one 32B B chunk
DEVI void stage_tile(uint32_t la0, uint32_t la1, uint32_t lb,
                     const bf16* ga0, const bf16* ga1, const bf16* gb) {
  async_b128<0 >(la0, ga0); async_b128<16>(la0, ga0);
  async_b128<32>(la0, ga0); async_b128<48>(la0, ga0);
  async_b128<0 >(la1, ga1); async_b128<16>(la1, ga1);
  async_b128<32>(la1, ga1); async_b128<48>(la1, ga1);
  async_b128<0 >(lb,  gb ); async_b128<16>(lb,  gb );
}

// ---------------- WMMA helpers (bf16 16x16x32, fp32 accum) ----------------
DEVI v8f wmma_bf16(v16bf a, v16bf b, v8f c) {
  return __builtin_amdgcn_wmma_f32_16x16x32_bf16(false, a, false, b,
                                                 (short)0, c, false, false);
}

// A tile in LDS as [rows][32 k] bf16. Frag: lanes 0-15 row=lane, K{0..15};
// lanes 16-31 row=lane-16, K{16..31} -> two 16B contiguous reads.
DEVI v16bf frag_a(const bf16* xs, int row0, int lane) {
  const int m  = row0 + (lane & 15);
  const int kb = (lane < 16) ? 0 : 16;
  const bf16* p = xs + m * 32 + kb;
  v8bf lo = *(const v8bf*)(p);
  v8bf hi = *(const v8bf*)(p + 8);
  v16bf a;
#pragma unroll
  for (int i = 0; i < 8; ++i) { a[i] = lo[i]; a[8 + i] = hi[i]; }
  return a;
}

// W tile transposed in LDS as [64 n][32 k]; B-frag: col n=lane&15, K contiguous
// (lanes<16: K0..15 else K16..31) = one 32B read.
DEVI v16bf frag_b(const bf16* wt, int n0, int lane) {
  const int n  = n0 + (lane & 15);
  const int kb = (lane < 16) ? 0 : 16;
  return *(const v16bf*)(wt + n * 32 + kb);
}

// ---------------- zero output + counters ----------------
__global__ void zero_kernel(float* __restrict__ out, int* __restrict__ cnt) {
  const size_t i = (size_t)blockIdx.x * blockDim.x + threadIdx.x;
  v4f z = {};
  ((v4f*)out)[i] = z;
  if (blockIdx.x == 0 && threadIdx.x < NEXP) cnt[threadIdx.x] = 0;
}

// ---------------- weight prep: fp32 [K][N] -> bf16 transposed [N][K] ----------------
__global__ __launch_bounds__(256) void wprep_kernel(
    const float* __restrict__ src, bf16* __restrict__ dst, int K, int N) {
  src += (size_t)blockIdx.z * K * N;
  dst += (size_t)blockIdx.z * K * N;
  const int k0 = blockIdx.x * 64;
  const int n0 = blockIdx.y * 64;
  const int t = threadIdx.x;
  __shared__ bf16 ts[64 * 72];
  {
    const int kl = t >> 2, cg = (t & 3) * 16;
    const float* sp = src + (size_t)(k0 + kl) * N + n0 + cg;
#pragma unroll
    for (int i4 = 0; i4 < 4; ++i4) {
      v4f v = *(const v4f*)(sp + i4 * 4);
#pragma unroll
      for (int i = 0; i < 4; ++i) ts[kl * 72 + cg + i4 * 4 + i] = (bf16)v[i];
    }
  }
  __syncthreads();
  {
    const int nl = t >> 2, kg = (t & 3) * 16;
    v16bf o;
#pragma unroll
    for (int i = 0; i < 16; ++i) o[i] = ts[(kg + i) * 72 + nl];
    *(v16bf*)(dst + (size_t)(n0 + nl) * K + k0 + kg) = o;
  }
}

// ---------------- gate: scores, softmax, top-3, routing lists, x->bf16 ----------------
__global__ __launch_bounds__(256) void gate_kernel(
    const float* __restrict__ x, const float* __restrict__ gw,
    const float* __restrict__ gb, float* __restrict__ weights,
    int* __restrict__ cnt, int* __restrict__ list, bf16* __restrict__ xbf) {
  const int lane = threadIdx.x & 31;
  const int wave = threadIdx.x >> 5;
  const int n    = blockIdx.x * 8 + wave;   // one wave32 per token

  float acc[NEXP] = {};
  const float* xr = x + (size_t)n * DIM;
  for (int d = lane; d < DIM; d += 32) {
    const float xv = xr[d];
    xbf[(size_t)n * DIM + d] = (bf16)xv;
    const float* g = gw + d * NEXP;
#pragma unroll
    for (int e = 0; e < NEXP; ++e) acc[e] += xv * g[e];
  }
#pragma unroll
  for (int e = 0; e < NEXP; ++e) {
#pragma unroll
    for (int off = 16; off > 0; off >>= 1)
      acc[e] += __shfl_xor(acc[e], off, 32);
    acc[e] += gb[e];
  }
  float mx = acc[0];
#pragma unroll
  for (int e = 1; e < NEXP; ++e) mx = fmaxf(mx, acc[e]);
  float p[NEXP], s = 0.f;
#pragma unroll
  for (int e = 0; e < NEXP; ++e) { p[e] = __expf(acc[e] - mx); s += p[e]; }
  int sel[TOPK];
  unsigned chosen = 0u;
#pragma unroll
  for (int k = 0; k < TOPK; ++k) {            // first-index tie-break
    float best = -3.4e38f; int bi = 0;
#pragma unroll
    for (int e = 0; e < NEXP; ++e)
      if (!((chosen >> e) & 1u) && acc[e] > best) { best = acc[e]; bi = e; }
    sel[k] = bi; chosen |= (1u << bi);
  }
  float psum = 0.f;
#pragma unroll
  for (int k = 0; k < TOPK; ++k) psum += p[sel[k]];
  const float denom = psum / s + 1e-8f;
  if (lane < NEXP) {
    float w = 0.f;
    if ((chosen >> lane) & 1u) w = (p[lane] / s) / denom;
    weights[(size_t)n * NEXP + lane] = w;
  }
  if (lane < TOPK) {
    const int es  = sel[lane];
    const int pos = atomicAdd(&cnt[es], 1);
    list[(size_t)es * N_TOK + pos] = n;
  }
}

// ---------------- expert GEMM 1: h = relu(x[list] @ w1 + b1) -> bf16 ----------------
// 128 threads = 4 waves; block tile 256(M) x 64(N); per-wave 64x64 (16 WMMA/iter);
// K-step 32; double-buffered LDS filled with async global->LDS copies.
__global__ __launch_bounds__(128) void mlp1_kernel(
    const bf16* __restrict__ xbf, const bf16* __restrict__ wTe,
    const float* __restrict__ b1e, const int* __restrict__ list,
    const int* __restrict__ cntp, bf16* __restrict__ hout) {
  const int count = *cntp;
  const int m0 = blockIdx.x * 256;
  if (m0 >= count) return;                    // uniform exit
  const int n0 = blockIdx.y * 64;
  const int t = threadIdx.x, lane = t & 31, wave = t >> 5;

  __shared__ __align__(64) bf16 xs[2][256 * 32];
  __shared__ __align__(64) bf16 wt[2][64 * 32];

  const int ra = min(m0 + 2 * t,     count - 1);
  const int rb = min(m0 + 2 * t + 1, count - 1);
  const bf16* aptr0 = xbf + (size_t)list[ra] * DIM;
  const bf16* aptr1 = xbf + (size_t)list[rb] * DIM;
  const bf16* bptr  = wTe + (size_t)(n0 + (t >> 1)) * DIM + (t & 1) * 16;

  const uint32_t la0[2] = { lds_off(&xs[0][(2 * t) * 32]),
                            lds_off(&xs[1][(2 * t) * 32]) };
  const uint32_t lb_[2] = { lds_off(&wt[0][(t >> 1) * 32 + (t & 1) * 16]),
                            lds_off(&wt[1][(t >> 1) * 32 + (t & 1) * 16]) };

  v8f acc[4][4] = {};
  float bias[4];
#pragma unroll
  for (int j = 0; j < 4; ++j) bias[j] = b1e[n0 + j * 16 + (lane & 15)];

  stage_tile(la0[0], la0[0] + 64, lb_[0], aptr0, aptr1, bptr);
  wait_async();
  __syncthreads();

  int buf = 0;
  for (int k0 = 0; k0 < DIM; k0 += 32, buf ^= 1) {
    const bool nxt = (k0 + 32) < DIM;
    if (nxt) {   // async-fill other buffer while computing this one
      stage_tile(la0[buf ^ 1], la0[buf ^ 1] + 64, lb_[buf ^ 1],
                 aptr0 + k0 + 32, aptr1 + k0 + 32, bptr + k0 + 32);
      if (k0 + 64 < DIM) {
        __builtin_prefetch(aptr0 + k0 + 64, 0, 2);
        __builtin_prefetch(bptr + k0 + 64, 0, 2);
      }
    }
    v16bf fa[4];
#pragma unroll
    for (int i = 0; i < 4; ++i) fa[i] = frag_a(xs[buf], wave * 64 + i * 16, lane);
#pragma unroll
    for (int j = 0; j < 4; ++j) {
      const v16bf fb = frag_b(wt[buf], j * 16, lane);
#pragma unroll
      for (int i = 0; i < 4; ++i) acc[i][j] = wmma_bf16(fa[i], fb, acc[i][j]);
    }
    if (nxt) wait_async();
    __syncthreads();
  }

#pragma unroll
  for (int i = 0; i < 4; ++i) {
#pragma unroll
    for (int r = 0; r < 8; ++r) {
      const int row = m0 + wave * 64 + i * 16 + r + ((lane >> 4) << 3);
      if (row < count) {
#pragma unroll
        for (int j = 0; j < 4; ++j) {
          float v = acc[i][j][r] + bias[j];
          v = v > 0.f ? v : 0.f;
          hout[(size_t)row * HID + (n0 + j * 16 + (lane & 15))] = (bf16)v;
        }
      }
    }
  }
}

// ---------------- expert GEMM 2: out[tok] += wgt * (h @ w2 + b2) ----------------
__global__ __launch_bounds__(128) void mlp2_kernel(
    const bf16* __restrict__ hbf, const bf16* __restrict__ wTe,
    const float* __restrict__ b2e, const int* __restrict__ list,
    const int* __restrict__ cntp, const float* __restrict__ weights,
    int e, float* __restrict__ out) {
  const int count = *cntp;
  const int m0 = blockIdx.x * 256;
  if (m0 >= count) return;
  const int n0 = blockIdx.y * 64;
  const int t = threadIdx.x, lane = t & 31, wave = t >> 5;

  __shared__ __align__(64) bf16 xs[2][256 * 32];
  __shared__ __align__(64) bf16 wt[2][64 * 32];

  const int ra = min(m0 + 2 * t,     count - 1);
  const int rb = min(m0 + 2 * t + 1, count - 1);
  const bf16* aptr0 = hbf + (size_t)ra * HID;
  const bf16* aptr1 = hbf + (size_t)rb * HID;
  const bf16* bptr  = wTe + (size_t)(n0 + (t >> 1)) * HID + (t & 1) * 16;

  const uint32_t la0[2] = { lds_off(&xs[0][(2 * t) * 32]),
                            lds_off(&xs[1][(2 * t) * 32]) };
  const uint32_t lb_[2] = { lds_off(&wt[0][(t >> 1) * 32 + (t & 1) * 16]),
                            lds_off(&wt[1][(t >> 1) * 32 + (t & 1) * 16]) };

  v8f acc[4][4] = {};
  float bias[4];
#pragma unroll
  for (int j = 0; j < 4; ++j) bias[j] = b2e[n0 + j * 16 + (lane & 15)];

  stage_tile(la0[0], la0[0] + 64, lb_[0], aptr0, aptr1, bptr);
  wait_async();
  __syncthreads();

  int buf = 0;
  for (int k0 = 0; k0 < HID; k0 += 32, buf ^= 1) {
    const bool nxt = (k0 + 32) < HID;
    if (nxt) {
      stage_tile(la0[buf ^ 1], la0[buf ^ 1] + 64, lb_[buf ^ 1],
                 aptr0 + k0 + 32, aptr1 + k0 + 32, bptr + k0 + 32);
      if (k0 + 64 < HID) {
        __builtin_prefetch(aptr0 + k0 + 64, 0, 2);
        __builtin_prefetch(bptr + k0 + 64, 0, 2);
      }
    }
    v16bf fa[4];
#pragma unroll
    for (int i = 0; i < 4; ++i) fa[i] = frag_a(xs[buf], wave * 64 + i * 16, lane);
#pragma unroll
    for (int j = 0; j < 4; ++j) {
      const v16bf fb = frag_b(wt[buf], j * 16, lane);
#pragma unroll
      for (int i = 0; i < 4; ++i) acc[i][j] = wmma_bf16(fa[i], fb, acc[i][j]);
    }
    if (nxt) wait_async();
    __syncthreads();
  }

#pragma unroll
  for (int i = 0; i < 4; ++i) {
#pragma unroll
    for (int r = 0; r < 8; ++r) {
      const int row = m0 + wave * 64 + i * 16 + r + ((lane >> 4) << 3);
      if (row < count) {
        const int tok = list[row];
        const float wgt = weights[(size_t)tok * NEXP + e];
#pragma unroll
        for (int j = 0; j < 4; ++j) {
          const int col = n0 + j * 16 + (lane & 15);
          out[(size_t)tok * OUTD + col] += wgt * (acc[i][j][r] + bias[j]);
        }
      }
    }
  }
}

// ---------------- host launch ----------------
extern "C" void kernel_launch(void* const* d_in, const int* in_sizes, int n_in,
                              void* d_out, int out_size, void* d_ws, size_t ws_size,
                              hipStream_t stream) {
  const float* x      = (const float*)d_in[0];
  const float* gate_w = (const float*)d_in[1];
  const float* gate_b = (const float*)d_in[2];
  const float* w1     = (const float*)d_in[3];
  const float* b1     = (const float*)d_in[4];
  const float* w2     = (const float*)d_in[5];
  const float* b2     = (const float*)d_in[6];
  float* out = (float*)d_out;

  char* ws = (char*)d_ws;
  size_t off = 0;
  auto carve = [&](size_t bytes) -> char* {
    char* p = ws + off;
    off = (off + bytes + 255) & ~(size_t)255;
    return p;
  };
  float* weights = (float*)carve((size_t)N_TOK * NEXP * sizeof(float));
  int*   cnt     = (int*)  carve((size_t)NEXP * sizeof(int));
  int*   list    = (int*)  carve((size_t)NEXP * N_TOK * sizeof(int));
  bf16*  xbf     = (bf16*) carve((size_t)N_TOK * DIM * sizeof(bf16));
  bf16*  hbf     = (bf16*) carve((size_t)N_TOK * HID * sizeof(bf16));
  bf16*  wT1     = (bf16*) carve((size_t)NEXP * DIM * HID * sizeof(bf16));
  bf16*  wT2     = (bf16*) carve((size_t)NEXP * HID * OUTD * sizeof(bf16));

  zero_kernel<<<dim3((N_TOK * OUTD) / (256 * 4)), 256, 0, stream>>>(out, cnt);
  wprep_kernel<<<dim3(DIM / 64, HID / 64, NEXP), 256, 0, stream>>>(w1, wT1, DIM, HID);
  wprep_kernel<<<dim3(HID / 64, OUTD / 64, NEXP), 256, 0, stream>>>(w2, wT2, HID, OUTD);
  gate_kernel<<<dim3(N_TOK / 8), 256, 0, stream>>>(x, gate_w, gate_b, weights,
                                                   cnt, list, xbf);
  for (int e = 0; e < NEXP; ++e) {
    mlp1_kernel<<<dim3(N_TOK / 256, HID / 64), 128, 0, stream>>>(
        xbf, wT1 + (size_t)e * DIM * HID, b1 + (size_t)e * HID,
        list + (size_t)e * N_TOK, cnt + e, hbf);
    mlp2_kernel<<<dim3(N_TOK / 256, OUTD / 64), 128, 0, stream>>>(
        hbf, wT2 + (size_t)e * HID * OUTD, b2 + (size_t)e * OUTD,
        list + (size_t)e * N_TOK, cnt + e, weights, e, out);
  }
}
